// Attention_24927990186686
// MI455X (gfx1250) — matrix operand order repaired and checked by reading, not compiled, not run on hardware
//
#include <hip/hip_runtime.h>
#include <hip/hip_bf16.h>

typedef __attribute__((ext_vector_type(16))) __bf16 v16bf;
typedef __attribute__((ext_vector_type(8)))  float  v8f;

#define N_TOT 4096
#define CHN   128
#define CK    64
#define WSTR  136   // LDS stride (bf16 elems) for W and xT tiles (272B: 16B-mult, bank-spread)
#define FSTR  72    // LDS stride for f key tile (144B)
#define HSTR  136   // LDS stride for h tile (272B)
#define PSTR  40    // LDS stride for p tile (80B)
#define MSTEP 128

static __device__ __forceinline__ unsigned short f32_bf16(float f) {
  unsigned u = __builtin_bit_cast(unsigned, f);
  u += 0x7FFFu + ((u >> 16) & 1u);          // round-to-nearest-even
  return (unsigned short)(u >> 16);
}
static __device__ __forceinline__ unsigned pack2(float a, float b) {
  return (unsigned)f32_bf16(a) | ((unsigned)f32_bf16(b) << 16);
}

union FragU { uint4 q[2]; v16bf v; };

// A/B fragments are both two 16B chunks: k = 8h..8h+7 and k = 16+8h..16+8h+7.
static __device__ __forceinline__ v16bf load_frag(const unsigned short* p0,
                                                  const unsigned short* p1) {
  FragU f;
  f.q[0] = *(const uint4*)p0;
  f.q[1] = *(const uint4*)p1;
  return f.v;
}

static __device__ __forceinline__ v8f wmma_bf16f32(v16bf a, v16bf b, v8f c) {
  // D = A(16x32 bf16) * B(32x16 bf16) + C(16x16 f32)
  return __builtin_amdgcn_wmma_f32_16x16x32_bf16(false, a, false, b, (short)0, c,
                                                 false, false);
}

// CDNA5 async DMA: global -> LDS, 16B per lane, tracked by ASYNCcnt.
static __device__ __forceinline__ void async_copy16(unsigned lds_addr,
                                                    const void* gptr) {
  asm volatile("global_load_async_to_lds_b128 %0, %1, off"
               :
               : "v"(lds_addr), "v"((unsigned long long)(size_t)gptr)
               : "memory");
}
static __device__ __forceinline__ void wait_async0() {
  asm volatile("s_wait_asynccnt 0x0" ::: "memory");
}

static __device__ __forceinline__ float redmax16(float v) {
  v = fmaxf(v, __shfl_xor(v, 1));
  v = fmaxf(v, __shfl_xor(v, 2));
  v = fmaxf(v, __shfl_xor(v, 4));
  v = fmaxf(v, __shfl_xor(v, 8));
  return v;
}
static __device__ __forceinline__ float redsum16(float v) {
  v += __shfl_xor(v, 1);
  v += __shfl_xor(v, 2);
  v += __shfl_xor(v, 4);
  v += __shfl_xor(v, 8);
  return v;
}

// ---------------------------------------------------------------------------
// Phase 1: 1x1-conv projections  f,g: [64x128] x x[b] ; h: [128x128] x x[b]
// Outputs bf16: fT,gT as [B][N][64] (n-major for phase-2 fragments),
//               h as [B][128][N] (c-major).
// ---------------------------------------------------------------------------
__global__ __launch_bounds__(256) void sagan_proj_kernel(
    const float* __restrict__ x,
    const float* __restrict__ Wf, const float* __restrict__ bfv,
    const float* __restrict__ Wg, const float* __restrict__ bgv,
    const float* __restrict__ Wh, const float* __restrict__ bhv,
    unsigned short* __restrict__ fT, unsigned short* __restrict__ gT,
    unsigned short* __restrict__ hB) {
  __shared__ unsigned short sW[256 * WSTR];   // Wf|Wg|Wh rows, bf16
  __shared__ unsigned short sXT[128 * WSTR];  // x tile transposed [m][k]
  __shared__ float sBias[256];

  const int b   = blockIdx.y;
  const int m0  = blockIdx.x * 128;
  const int tid = threadIdx.x;

  // Stage all 256 weight rows (bf16) + bias once per block.
  for (int idx = tid; idx < 256 * 128; idx += 256) {
    int r = idx >> 7, c = idx & 127;
    float w = (r < 64) ? Wf[r * 128 + c]
                       : (r < 128) ? Wg[(r - 64) * 128 + c]
                                   : Wh[(r - 128) * 128 + c];
    sW[r * WSTR + c] = f32_bf16(w);
  }
  sBias[tid] = (tid < 64) ? bfv[tid] : (tid < 128) ? bgv[tid - 64] : bhv[tid - 128];

  // Stage x tile transposed to [m][k] bf16 (coalesced reads along m).
  const float* xb = x + (size_t)b * CHN * N_TOT;
  for (int idx = tid; idx < 128 * 128; idx += 256) {
    int k = idx >> 7, m = idx & 127;
    sXT[m * WSTR + k] = f32_bf16(xb[(size_t)k * N_TOT + m0 + m]);
  }
  __syncthreads();

  const int lane = tid & 31;
  const int wave = tid >> 5;
  const int hf   = lane >> 4;     // lane half
  const int l16  = lane & 15;
  const int koff = hf ? 8 : 0;    // A-fragment K sub-offset
  const int mloc = wave * 16 + l16;
  const int m    = m0 + mloc;

  // B fragments: xT row mloc, 4 K-chunks of 32.
  v16bf bxf[4];
  const unsigned short* xrow = sXT + mloc * WSTR;
#pragma unroll
  for (int kc = 0; kc < 4; ++kc) {
    const unsigned short* p = xrow + kc * 32 + hf * 8;
    bxf[kc] = load_frag(p, p + 16);
  }

  for (int ct = 0; ct < 16; ++ct) {   // 4 f-tiles, 4 g-tiles, 8 h-tiles
    v8f acc;
#pragma unroll
    for (int r = 0; r < 8; ++r) acc[r] = 0.f;
    const unsigned short* wrow = sW + (ct * 16 + l16) * WSTR;
#pragma unroll
    for (int kc = 0; kc < 4; ++kc) {
      const unsigned short* p = wrow + kc * 32;
      acc = wmma_bf16f32(load_frag(p + koff, p + koff + 16), bxf[kc], acc);
    }
    const int cbase = ct * 16 + hf * 8;
#pragma unroll
    for (int r = 0; r < 8; ++r) acc[r] += sBias[cbase + r];

    if (ct < 8) {  // fT / gT : [B][N][64], 16B packed store per lane
      unsigned short* dst = (ct < 4 ? fT : gT) +
                            ((size_t)b * N_TOT + m) * CK + (ct & 3) * 16 + hf * 8;
      uint4 q;
      q.x = pack2(acc[0], acc[1]);
      q.y = pack2(acc[2], acc[3]);
      q.z = pack2(acc[4], acc[5]);
      q.w = pack2(acc[6], acc[7]);
      *(uint4*)dst = q;
    } else {       // h : [B][128][N]
      unsigned short* hrow =
          hB + ((size_t)b * CHN + (ct - 8) * 16 + hf * 8) * N_TOT + m;
#pragma unroll
      for (int r = 0; r < 8; ++r) hrow[(size_t)r * N_TOT] = f32_bf16(acc[r]);
    }
  }
}

// ---------------------------------------------------------------------------
// Phase 2: flash attention.  s = gT·f (K=64), online softmax over m,
// o += p·hT (K=m), out = gamma*o/l + x.  Wave owns 16 query rows.
// Key/value tiles are double-buffered in LDS and filled with CDNA5 async
// global->LDS DMA so the fill of tile i+1 overlaps the 48 WMMAs of tile i.
// ---------------------------------------------------------------------------
__global__ __launch_bounds__(256) void sagan_attn_kernel(
    const unsigned short* __restrict__ fT, const unsigned short* __restrict__ gT,
    const unsigned short* __restrict__ hB, const float* __restrict__ x,
    const float* __restrict__ gamma, float* __restrict__ out) {
  __shared__ unsigned short sF[2][MSTEP * FSTR];  // key tiles   [m][64]
  __shared__ unsigned short sH[2][CHN * HSTR];    // value tiles [c][m]
  __shared__ unsigned short sP[8][16 * PSTR];     // per-wave prob tile [n][32]

  const int b    = blockIdx.y;
  const int n0   = blockIdx.x * 128;
  const int tid  = threadIdx.x;
  const int lane = tid & 31;
  const int wave = tid >> 5;
  const int hf   = lane >> 4;
  const int l16  = lane & 15;
  const int koff = hf ? 8 : 0;
  const int nw   = n0 + wave * 16;

  const unsigned short* fTb = fT + (size_t)b * N_TOT * CK;
  const unsigned short* hBb = hB + (size_t)b * CHN * N_TOT;

  // Issue async DMA for one 128-key/value tile into LDS buffer `bi`.
  auto stage_async = [&](int bi, int ms) {
    for (int idx = tid; idx < MSTEP * (CK / 8); idx += 256) {
      int mrow = idx >> 3, w = idx & 7;
      async_copy16((unsigned)(size_t)(sF[bi] + mrow * FSTR + w * 8),
                   fTb + (size_t)(ms + mrow) * CK + w * 8);
    }
    for (int idx = tid; idx < CHN * (MSTEP / 8); idx += 256) {
      int c = idx >> 4, w = idx & 15;
      async_copy16((unsigned)(size_t)(sH[bi] + c * HSTR + w * 8),
                   hBb + (size_t)c * N_TOT + ms + w * 8);
    }
  };

  // Query A-fragments: gT rows nw..nw+15, K = 64 -> two chunks.
  const unsigned short* grow = gT + ((size_t)b * N_TOT + nw + l16) * CK;
  const v16bf ga0 = load_frag(grow + koff, grow + koff + 16);
  const v16bf ga1 = load_frag(grow + 32 + koff, grow + 32 + koff + 16);

  v8f oacc[8];
#pragma unroll
  for (int t = 0; t < 8; ++t)
#pragma unroll
    for (int r = 0; r < 8; ++r) oacc[t][r] = 0.f;

  float mmax[8], lsum[8];
#pragma unroll
  for (int r = 0; r < 8; ++r) { mmax[r] = -__builtin_inff(); lsum[r] = 0.f; }

  stage_async(0, 0);  // prologue fill of buffer 0
  int buf = 0;

  for (int ms = 0; ms < N_TOT; ms += MSTEP) {
    wait_async0();     // this wave's DMA into `buf` complete
    __syncthreads();   // -> every wave's DMA complete; prev reads of buf^1 done

    if (ms + MSTEP < N_TOT) stage_async(buf ^ 1, ms + MSTEP);  // overlap fill
    if (ms + 2 * MSTEP < N_TOT) {  // warm L2 one more tile ahead
      __builtin_prefetch(fTb + (size_t)(ms + 2 * MSTEP) * CK + tid * 32, 0, 1);
      __builtin_prefetch(hBb + (size_t)(tid >> 1) * N_TOT + ms + 2 * MSTEP +
                             (tid & 1) * 64, 0, 1);
    }
    const unsigned short* cF = sF[buf];
    const unsigned short* cH = sH[buf];

    for (int mt = 0; mt < MSTEP; mt += 32) {
      // --- s tiles: two 16x16 score tiles over K = 64 ---
      v8f s0, s1;
#pragma unroll
      for (int r = 0; r < 8; ++r) { s0[r] = 0.f; s1[r] = 0.f; }
#pragma unroll
      for (int kc = 0; kc < 2; ++kc) {
        const v16bf a = kc ? ga1 : ga0;
        const unsigned short* p0 = cF + (mt + l16) * FSTR + kc * 32 + hf * 8;
        const unsigned short* p1 = cF + (mt + 16 + l16) * FSTR + kc * 32 + hf * 8;
        s0 = wmma_bf16f32(a, load_frag(p0, p0 + 16), s0);
        s1 = wmma_bf16f32(a, load_frag(p1, p1 + 16), s1);
      }

      // --- online softmax (row r of D lives across the 16 lanes of a half) ---
      float alpha[8];
#pragma unroll
      for (int r = 0; r < 8; ++r) {
        float v  = redmax16(fmaxf(s0[r], s1[r]));
        float nm = fmaxf(mmax[r], v);
        alpha[r] = __expf(mmax[r] - nm);
        mmax[r]  = nm;
        float e0 = __expf(s0[r] - nm);
        float e1 = __expf(s1[r] - nm);
        s0[r] = e0; s1[r] = e1;
        lsum[r] = lsum[r] * alpha[r] + redsum16(e0 + e1);
      }
#pragma unroll
      for (int t = 0; t < 8; ++t)
#pragma unroll
        for (int r = 0; r < 8; ++r) oacc[t][r] *= alpha[r];

      // --- D-layout -> A-layout transpose of p via per-wave LDS tile ---
      unsigned short* pw = &sP[wave][0];
#pragma unroll
      for (int r = 0; r < 8; ++r) {
        int row = hf * 8 + r;
        pw[row * PSTR + l16]      = f32_bf16(s0[r]);
        pw[row * PSTR + 16 + l16] = f32_bf16(s1[r]);
      }
      const unsigned short* pr = pw + l16 * PSTR;   // same-wave, DS in-order
      const v16bf pa = load_frag(pr + koff, pr + koff + 16);

      // --- o accumulate: 8 channel tiles, K = 32 keys ---
#pragma unroll
      for (int ct = 0; ct < 8; ++ct) {
        const unsigned short* hp = cH + (ct * 16 + l16) * HSTR + mt + hf * 8;
        oacc[ct] = wmma_bf16f32(pa, load_frag(hp, hp + 16), oacc[ct]);
      }
    }
    buf ^= 1;
  }

  // --- finalize: out = gamma * o / l + x  (32B coalesced along n) ---
  const float gm = gamma[0];
  float inv[8];
#pragma unroll
  for (int r = 0; r < 8; ++r) inv[r] = 1.0f / lsum[r];
  const float* xb = x + (size_t)b * CHN * N_TOT;
  float* ob = out + (size_t)b * CHN * N_TOT;
#pragma unroll
  for (int ct = 0; ct < 8; ++ct) {
    const int c = ct * 16 + l16;
    const size_t base = (size_t)c * N_TOT + nw + hf * 8;
#pragma unroll
    for (int r = 0; r < 8; ++r)
      ob[base + r] = gm * (oacc[ct][r] * inv[r]) + xb[base + r];
  }
}

// ---------------------------------------------------------------------------
extern "C" void kernel_launch(void* const* d_in, const int* in_sizes, int n_in,
                              void* d_out, int out_size, void* d_ws, size_t ws_size,
                              hipStream_t stream) {
  const float* x     = (const float*)d_in[0];
  const float* Wf    = (const float*)d_in[1];
  const float* bfv   = (const float*)d_in[2];
  const float* Wg    = (const float*)d_in[3];
  const float* bgv   = (const float*)d_in[4];
  const float* Wh    = (const float*)d_in[5];
  const float* bhv   = (const float*)d_in[6];
  const float* gamma = (const float*)d_in[7];
  float* out = (float*)d_out;

  // Workspace carve (bf16): fT 4MB | gT 4MB | h 8MB  (16MB total)
  unsigned short* fT = (unsigned short*)d_ws;
  unsigned short* gT = fT + (size_t)8 * N_TOT * CK;
  unsigned short* hB = gT + (size_t)8 * N_TOT * CK;

  dim3 grid(32, 8);  // 32 pixel/query chunks x 8 batches
  sagan_proj_kernel<<<grid, 256, 0, stream>>>(x, Wf, bfv, Wg, bgv, Wh, bhv,
                                              fT, gT, hB);
  sagan_attn_kernel<<<grid, 256, 0, stream>>>(fT, gT, hB, x, gamma, out);
}
